// PairStack_47278999994726
// MI455X (gfx1250) — matrix operand order repaired and checked by reading, not compiled, not run on hardware
//
#include <hip/hip_runtime.h>
#include <hip/hip_bf16.h>
#include <math.h>

// ---------------------------------------------------------------------------
// Types / constants
// ---------------------------------------------------------------------------
typedef __bf16 bf16;
typedef __attribute__((ext_vector_type(16))) __bf16 v16bf;
typedef __attribute__((ext_vector_type(8)))  __bf16 v8bf;
typedef __attribute__((ext_vector_type(8)))  float  v8f;
typedef __attribute__((ext_vector_type(4)))  unsigned int u32x4;
typedef __attribute__((ext_vector_type(8)))  int          i32x8;
typedef __attribute__((ext_vector_type(4)))  int          i32x4;

#define NSEQ   256
#define NPAIR  (NSEQ * NSEQ)      // 65536 pair rows
#define CZD    128
#define HATT   4
#define CAH    32
#define TNCZ   512                // TN * CZ
#define LN_EPS 1e-5f
#define INF_M  1e8f

__device__ __forceinline__ float dsigmoid(float x) { return 1.0f / (1.0f + __expf(-x)); }

__device__ __forceinline__ float waveRedSum(float v) {
    v += __shfl_xor(v, 16, 32);
    v += __shfl_xor(v, 8, 32);
    v += __shfl_xor(v, 4, 32);
    v += __shfl_xor(v, 2, 32);
    v += __shfl_xor(v, 1, 32);
    return v;
}

// Low 32 bits of a generic pointer to LDS == wave-relative LDS byte offset
// (flat LDS aperture places the offset in addr[31:0], ISA 10.2).
__device__ __forceinline__ unsigned ldsOff(const void* p) {
    return (unsigned)(unsigned long long)p;
}

// ---------------------------------------------------------------------------
// TDM: stage a 2D tile (rows x 32 bf16 elems, row stride 128 elems) to LDS.
// D# layout per CDNA5 ISA (group0 128b, group1 256b).
// ---------------------------------------------------------------------------
#if __has_builtin(__builtin_amdgcn_tensor_load_to_lds)
#define HAVE_TDM 1
__device__ __forceinline__ void tdm_load_tile_256x32(unsigned lds, const void* gbase)
{
    unsigned long long ga = (unsigned long long)gbase;
    u32x4 g0;
    g0[0] = 1u;                                   // count=1, user mode
    g0[1] = lds;                                  // lds_addr
    g0[2] = (unsigned)(ga & 0xFFFFFFFFu);         // global_addr[31:0]
    g0[3] = (unsigned)((ga >> 32) & 0x1FFFFFFu)   // global_addr[56:32]
          | (2u << 30);                           // type=2 ("image")
    i32x8 g1;
    g1[0] = (int)(1u << 16);                      // data_size=1 (2 bytes)
    g1[1] = (int)(32u << 16);                     // tensor_dim0=32 (bits 63:48)
    g1[2] = (int)(256u << 16);                    // tensor_dim1=256 (bits 95:80..)
    g1[3] = (int)(32u << 16);                     // tile_dim0=32 (bits 127:112)
    g1[4] = (int)(256u);                          // tile_dim1=256, tile_dim2=0
    g1[5] = (int)(128u);                          // tensor_dim0_stride=128 elems
    g1[6] = 0;                                    // stride0 hi / stride1 lo
    g1[7] = 0;
    i32x4 z4 = {0, 0, 0, 0};
#if defined(__clang_major__) && (__clang_major__ >= 23)
    i32x8 z8 = {0, 0, 0, 0, 0, 0, 0, 0};
    __builtin_amdgcn_tensor_load_to_lds(g0, g1, z4, z4, z8, 0);
#else
    __builtin_amdgcn_tensor_load_to_lds(g0, g1, z4, z4, 0);
#endif
}
#endif

__device__ __forceinline__ void waitTensor0()
{
#if __has_builtin(__builtin_amdgcn_s_wait_tensorcnt)
    __builtin_amdgcn_s_wait_tensorcnt(0);
#else
    asm volatile("s_wait_tensorcnt 0x0" ::: "memory");
#endif
}

__device__ __forceinline__ void waitAsync0()
{
#if __has_builtin(__builtin_amdgcn_s_wait_asynccnt)
    __builtin_amdgcn_s_wait_asynccnt(0);
#else
    asm volatile("s_wait_asynccnt 0x0" ::: "memory");
#endif
}

// Async stage: one 256x32 bf16 tile (16KB) -> LDS, 8 lines x 128 thr x 16B.
__device__ __forceinline__ void asyncStage_256x32(unsigned ldsBase, const bf16* gbase,
                                                  size_t rowStrideElems, int tid)
{
#pragma unroll
    for (int line = 0; line < 8; ++line) {
        int u   = line * 128 + tid;               // 16-byte unit
        int k   = u >> 2;                         // row
        int sub = u & 3;                          // 16B chunk within 64B row
        const char* g = (const char*)(gbase + (size_t)k * rowStrideElems) + sub * 16;
        unsigned l = ldsBase + (unsigned)u * 16u;
        asm volatile("global_load_async_to_lds_b128 %0, %1, off"
                     :: "v"(l), "v"(g) : "memory");
    }
}

// ---------------------------------------------------------------------------
// LayerNorm over 128 channels, one wave per pair row, output bf16.
// ---------------------------------------------------------------------------
__global__ __launch_bounds__(256)
void ln_bf16_kernel(const float* __restrict__ src, const float* __restrict__ gam,
                    const float* __restrict__ bet, bf16* __restrict__ dst,
                    int transposeRows)
{
    int wave = (int)((blockIdx.x * blockDim.x + threadIdx.x) >> 5);
    int lane = threadIdx.x & 31;
    if (wave >= NPAIR) return;
    const float4 x = ((const float4*)(src + (size_t)wave * CZD))[lane];
    float s = x.x + x.y + x.z + x.w;
    float q = x.x * x.x + x.y * x.y + x.z * x.z + x.w * x.w;
    s = waveRedSum(s);
    q = waveRedSum(q);
    float mean = s * (1.0f / CZD);
    float var  = q * (1.0f / CZD) - mean * mean;
    float rstd = rsqrtf(var + LN_EPS);
    int outRow = wave;
    if (transposeRows) { int i = wave >> 8, j = wave & 255; outRow = (j << 8) | i; }
    bf16* orow = dst + (size_t)outRow * CZD;
    int c = lane * 4;
    orow[c + 0] = (bf16)((x.x - mean) * rstd * gam[c + 0] + bet[c + 0]);
    orow[c + 1] = (bf16)((x.y - mean) * rstd * gam[c + 1] + bet[c + 1]);
    orow[c + 2] = (bf16)((x.z - mean) * rstd * gam[c + 2] + bet[c + 2]);
    orow[c + 3] = (bf16)((x.w - mean) * rstd * gam[c + 3] + bet[c + 3]);
}

// ---------------------------------------------------------------------------
// Weight prep: f32 W[K][N] row-major  ->  bf16 Wt[N][K] (K contiguous)
// ---------------------------------------------------------------------------
__global__ void wprep_kernel(const float* __restrict__ w, bf16* __restrict__ wt,
                             int K, int Nout)
{
    int idx = blockIdx.x * blockDim.x + threadIdx.x;
    if (idx >= K * Nout) return;
    int n = idx / K, k = idx - n * K;
    wt[idx] = (bf16)w[(size_t)k * Nout + n];
}

// ---------------------------------------------------------------------------
// Generic WMMA bf16 GEMM:  C[M,Nout] = A[M,K] @ Bt[Nout,K]^T  (NT form)
// Block = 256 threads = 8 waves; block tile 64(M) x 32(N); wave tile 16x16.
// grid = (Nout/32, M/64, batch)
// Epilogues (epi):
//   0: outF = acc+bias                 1: outF = sigmoid(acc+bias)
//   2: outH = bf16(acc+bias)           3: outH = bf16(silu(acc+bias))
//   4: addOut[idx] += acc+bias         5: outH = bf16((acc+bias)*scale)
// ---------------------------------------------------------------------------
__global__ __launch_bounds__(256)
void gemm_wmma_kernel(const bf16* __restrict__ A, const bf16* __restrict__ Bt,
                      const float* __restrict__ bias,
                      int M, int K, int Nout, int epi, float scale,
                      float* __restrict__ outF, bf16* __restrict__ outH,
                      float* __restrict__ addOut,
                      long long outRowStride, long long outColStride,
                      long long outOffset, int transposeRows,
                      long long aBatch, long long bBatch, long long oBatch)
{
    const bf16* Ab  = A  + (size_t)blockIdx.z * aBatch;
    const bf16* Btb = Bt + (size_t)blockIdx.z * bBatch;
    long long   ooff = outOffset + (long long)blockIdx.z * oBatch;

    int w    = threadIdx.x >> 5;
    int lane = threadIdx.x & 31;
    int lr   = lane & 15;
    int hi   = lane >> 4;                    // which half-wave
    int row0 = blockIdx.y * 64 + (w & 3) * 16;
    int col0 = blockIdx.x * 32 + (w >> 2) * 16;

    const bf16* aRow = Ab  + (size_t)(row0 + lr) * K;
    const bf16* bRow = Btb + (size_t)(col0 + lr) * K;
    int aDelta = hi * 8;                     // A: lanes>=16 hold K+8 .. (ISA layout)
    int bDelta = hi * 8;                     // B: lanes>=16 hold K+8 .. (same as A)

    v8f acc = {};
    for (int k0 = 0; k0 < K; k0 += 32) {
        __builtin_prefetch(aRow + k0 + 128, 0, 1);
        __builtin_prefetch(bRow + k0 + 128, 0, 1);
        v8bf a0 = *(const v8bf*)(aRow + k0 + aDelta);
        v8bf a1 = *(const v8bf*)(aRow + k0 + 16 + aDelta);
        v8bf b0 = *(const v8bf*)(bRow + k0 + bDelta);
        v8bf b1 = *(const v8bf*)(bRow + k0 + 16 + bDelta);
        v16bf av, bv;
#pragma unroll
        for (int t = 0; t < 8; ++t) {
            av[t] = a0[t]; av[t + 8] = a1[t];
            bv[t] = b0[t]; bv[t + 8] = b1[t];
        }
        acc = __builtin_amdgcn_wmma_f32_16x16x32_bf16(false, av, false, bv,
                                                      (short)0, acc, false, false);
    }

    int   nCol = col0 + lr;
    float bval = bias ? bias[nCol] : 0.0f;
#pragma unroll
    for (int r = 0; r < 8; ++r) {
        int m = row0 + r + hi * 8;           // C layout: VGPR r -> M=r / M=8+r
        float v = acc[r] + bval;
        long long rr = m;
        if (transposeRows) { int i = m >> 8, j = m & 255; rr = (j << 8) | i; }
        long long idx = rr * outRowStride + (long long)nCol * outColStride + ooff;
        switch (epi) {
            case 0: outF[idx] = v; break;
            case 1: outF[idx] = dsigmoid(v); break;
            case 2: outH[idx] = (bf16)v; break;
            case 3: outH[idx] = (bf16)(v * dsigmoid(v)); break;
            case 4: addOut[idx] += v; break;
            case 5: outH[idx] = (bf16)(v * scale); break;
        }
    }
}

// ---------------------------------------------------------------------------
// Triangle-attention: one block per (jblock, head, i); 128 threads = 4 waves.
// K tile staged via TDM (tensor_load_to_lds), V tile via async-to-LDS DMA;
// S = Q K^T (+bias+mask) -> softmax (probs overwrite S rows as bf16) -> P V.
// LDS: S 32KB + KL 16KB + VL 16KB = 64KB.
// ---------------------------------------------------------------------------
__global__ __launch_bounds__(128)
void attn_kernel(const bf16* __restrict__ Q, const bf16* __restrict__ Km,
                 const bf16* __restrict__ Vm, const float* __restrict__ tb,
                 const float* __restrict__ mask, int maskT,
                 float* __restrict__ outO)
{
    __shared__ float S[32][256];
    __shared__ bf16  KL[256][32];
    __shared__ bf16  VL[256][32];
    int jb = blockIdx.x;                     // 0..7  (32-row j tile)
    int h  = blockIdx.y;                     // 0..3
    int i  = blockIdx.z;                     // 0..255
    int w    = threadIdx.x >> 5;
    int lane = threadIdx.x & 31;
    int lr   = lane & 15;
    int hi   = lane >> 4;

    const bf16* kBase = Km + ((size_t)i * 256) * CZD + h * CAH;
    const bf16* vBase = Vm + ((size_t)i * 256) * CZD + h * CAH;

    // -------- stage V tile (async DMA, consumed in phase 3) --------
    asyncStage_256x32(ldsOff(&VL[0][0]), vBase, CZD, (int)threadIdx.x);

    // -------- stage K tile (Tensor Data Mover; wave 0 issues) --------
#ifdef HAVE_TDM
    if (w == 0) tdm_load_tile_256x32(ldsOff(&KL[0][0]), kBase);
    waitTensor0();
#else
    asyncStage_256x32(ldsOff(&KL[0][0]), kBase, CZD, (int)threadIdx.x);
    waitAsync0();
#endif
    __syncthreads();

    // -------- phase 1: scores (K from LDS) --------
    {
        int jt = w & 1;
        int j0 = jb * 32 + jt * 16;
        const bf16* qRow = Q + ((size_t)(i * 256 + j0 + lr) * CZD + h * CAH);
        v8bf a0 = *(const v8bf*)(qRow + hi * 8);
        v8bf a1 = *(const v8bf*)(qRow + 16 + hi * 8);
        v16bf av;
#pragma unroll
        for (int t = 0; t < 8; ++t) { av[t] = a0[t]; av[t + 8] = a1[t]; }
        for (int tt = 0; tt < 8; ++tt) {
            int kt = (w >> 1) + 2 * tt;
            int k0 = kt * 16;
            const bf16* kRow = &KL[k0 + lr][0] + hi * 8;
            v8bf b0 = *(const v8bf*)(kRow);
            v8bf b1 = *(const v8bf*)(kRow + 16);
            v16bf bv;
#pragma unroll
            for (int t = 0; t < 8; ++t) { bv[t] = b0[t]; bv[t + 8] = b1[t]; }
            v8f acc = {};
            acc = __builtin_amdgcn_wmma_f32_16x16x32_bf16(false, av, false, bv,
                                                          (short)0, acc, false, false);
#pragma unroll
            for (int r = 0; r < 8; ++r) {
                int m  = r + hi * 8;
                int jg = j0 + m;
                int kg = k0 + lr;
                float mval = maskT ? mask[kg * 256 + i] : mask[i * 256 + kg];
                float sv = acc[r] + tb[((size_t)jg * 256 + kg) * HATT + h]
                                  + INF_M * (mval - 1.0f);
                S[jt * 16 + m][kg] = sv;
            }
        }
    }
    __syncthreads();

    // -------- phase 2: softmax over k; probs overwrite S rows as bf16 ------
    if (threadIdx.x < 32) {
        int r = threadIdx.x;
        float mx = -3.4e38f;
        for (int k = 0; k < 256; ++k) mx = fmaxf(mx, S[r][k]);
        float sum = 0.0f;
        for (int k = 0; k < 256; ++k) { float e = __expf(S[r][k] - mx); S[r][k] = e; sum += e; }
        float inv = 1.0f / sum;
        bf16* Pr = (bf16*)&S[r][0];          // in-place: write@2k < read@4k
        for (int k = 0; k < 256; ++k) Pr[k] = (bf16)(S[r][k] * inv);
    }
    waitAsync0();                            // V tile must be in LDS past here
    __syncthreads();

    // -------- phase 3: O = P @ V (both operands from LDS) --------
    {
        int jt = w & 1;
        int ct = w >> 1;                     // CA=32 -> two 16-col tiles
        int jl = jt * 16 + lr;
        const bf16* Pj = (const bf16*)&S[jl][0];
        v8f acc = {};
        for (int kb = 0; kb < 8; ++kb) {
            int k0 = kb * 32;
            v8bf a0 = *(const v8bf*)(Pj + k0 + hi * 8);
            v8bf a1 = *(const v8bf*)(Pj + k0 + 16 + hi * 8);
            v16bf av;
#pragma unroll
            for (int t = 0; t < 8; ++t) { av[t] = a0[t]; av[t + 8] = a1[t]; }
            v16bf bv;
            int c     = ct * 16 + lr;
            int kbase = k0 + hi * 8;
#pragma unroll
            for (int t = 0; t < 16; ++t) bv[t] = VL[kbase + t + (t >= 8 ? 8 : 0)][c];
            acc = __builtin_amdgcn_wmma_f32_16x16x32_bf16(false, av, false, bv,
                                                          (short)0, acc, false, false);
        }
#pragma unroll
        for (int r = 0; r < 8; ++r) {
            int m  = r + hi * 8;
            int jg = jb * 32 + jt * 16 + m;
            outO[((size_t)(i * 256 + jg)) * CZD + h * CAH + ct * 16 + lr] = acc[r];
        }
    }
}

// ---------------------------------------------------------------------------
// Pair-bias projection: tbout[row][h] = sum_c zn[row][c] * tbw[c][h]
// ---------------------------------------------------------------------------
__global__ void tbproj_kernel(const bf16* __restrict__ zn, const float* __restrict__ tbw,
                              float* __restrict__ out)
{
    int idx = blockIdx.x * blockDim.x + threadIdx.x;
    if (idx >= NPAIR * HATT) return;
    int row = idx >> 2, h = idx & 3;
    const bf16* zr = zn + (size_t)row * CZD;
    float s = 0.0f;
#pragma unroll 8
    for (int c = 0; c < CZD; ++c) s += (float)zr[c] * tbw[c * HATT + h];
    out[idx] = s;
}

// ---------------------------------------------------------------------------
// Elementwise kernels
// ---------------------------------------------------------------------------
__global__ void gatepack_kernel(const float* __restrict__ sig, const float* __restrict__ proj,
                                const float* __restrict__ mask, bf16* __restrict__ out_cmaj,
                                int transposePair)
{
    int e = blockIdx.x * blockDim.x + threadIdx.x;
    if (e >= NPAIR * CZD) return;
    int row = e >> 7, c = e & 127;
    int p = row >> 8, q = row & 255;
    float v = mask[row] * sig[e] * proj[e];
    int pr = transposePair ? ((q << 8) | p) : row;
    out_cmaj[(size_t)c * NPAIR + pr] = (bf16)v;
}

__global__ void fma_add_kernel(float* __restrict__ z, const float* __restrict__ a,
                               const float* __restrict__ b, int n)
{
    int e = blockIdx.x * blockDim.x + threadIdx.x;
    if (e < n) z[e] += a[e] * b[e];
}

__global__ void mul_f32_to_bf16_kernel(const float* __restrict__ a, const float* __restrict__ b,
                                       bf16* __restrict__ o, int n)
{
    int e = blockIdx.x * blockDim.x + threadIdx.x;
    if (e < n) o[e] = (bf16)(a[e] * b[e]);
}

__global__ void mul_bf16_ip_kernel(bf16* __restrict__ a, const bf16* __restrict__ b, int n)
{
    int e = blockIdx.x * blockDim.x + threadIdx.x;
    if (e < n) a[e] = (bf16)((float)a[e] * (float)b[e]);
}

// ---------------------------------------------------------------------------
// Host-side parameter bundles
// ---------------------------------------------------------------------------
struct TriMulP  { const float *ln_g,*ln_b,*ag_w,*ag_b,*ap_w,*ap_b,*bg_w,*bg_b,
                              *bp_w,*bp_b,*lno_g,*lno_b,*o_w,*o_b,*g_w,*g_b; };
struct TriAttnP { const float *ln_g,*ln_b,*q_w,*k_w,*v_w,*tb_w,*g_w,*g_b,*o_w,*o_b; };
struct TransP   { const float *ln_g,*ln_b,*a_w,*b_w,*out_w; };

extern "C" void kernel_launch(void* const* d_in, const int* in_sizes, int n_in,
                              void* d_out, int out_size, void* d_ws, size_t ws_size,
                              hipStream_t stream)
{
    (void)n_in; (void)out_size; (void)ws_size;
    // ---------------- input mapping (handles both flatten orders) ----------
    const float* zin  = nullptr;
    const float* mask = nullptr;
    TriMulP tmo{}, tmi{}; TriAttnP tas{}, tae{}; TransP trs{};
    int cur = 0;
    auto nxt = [&]() { return (const float*)d_in[cur++]; };
    if (in_sizes[0] == NPAIR * CZD) {
        // insertion order: z, pair_mask, tmo, tmi, tas, tae, trans
        zin = nxt(); mask = nxt();
        auto rdTM = [&](TriMulP& p){ p.ln_g=nxt(); p.ln_b=nxt(); p.ag_w=nxt(); p.ag_b=nxt();
            p.ap_w=nxt(); p.ap_b=nxt(); p.bg_w=nxt(); p.bg_b=nxt(); p.bp_w=nxt(); p.bp_b=nxt();
            p.lno_g=nxt(); p.lno_b=nxt(); p.o_w=nxt(); p.o_b=nxt(); p.g_w=nxt(); p.g_b=nxt(); };
        auto rdTA = [&](TriAttnP& p){ p.ln_g=nxt(); p.ln_b=nxt(); p.q_w=nxt(); p.k_w=nxt();
            p.v_w=nxt(); p.tb_w=nxt(); p.g_w=nxt(); p.g_b=nxt(); p.o_w=nxt(); p.o_b=nxt(); };
        auto rdTR = [&](TransP& p){ p.ln_g=nxt(); p.ln_b=nxt(); p.a_w=nxt(); p.b_w=nxt(); p.out_w=nxt(); };
        rdTM(tmo); rdTM(tmi); rdTA(tas); rdTA(tae); rdTR(trs);
    } else {
        // sorted pytree order: pair_mask, tae, tas, tmi, tmo, trans, z
        mask = nxt();
        auto rdTA = [&](TriAttnP& p){ p.g_b=nxt(); p.g_w=nxt(); p.k_w=nxt(); p.ln_b=nxt();
            p.ln_g=nxt(); p.o_b=nxt(); p.o_w=nxt(); p.q_w=nxt(); p.tb_w=nxt(); p.v_w=nxt(); };
        auto rdTM = [&](TriMulP& p){ p.ag_b=nxt(); p.ag_w=nxt(); p.ap_b=nxt(); p.ap_w=nxt();
            p.bg_b=nxt(); p.bg_w=nxt(); p.bp_b=nxt(); p.bp_w=nxt(); p.g_b=nxt(); p.g_w=nxt();
            p.ln_b=nxt(); p.ln_g=nxt(); p.lno_b=nxt(); p.lno_g=nxt(); p.o_b=nxt(); p.o_w=nxt(); };
        auto rdTR = [&](TransP& p){ p.a_w=nxt(); p.b_w=nxt(); p.ln_b=nxt(); p.ln_g=nxt(); p.out_w=nxt(); };
        rdTA(tae); rdTA(tas); rdTM(tmi); rdTM(tmo); rdTR(trs);
        zin = nxt();
    }

    // ---------------- workspace layout ----------------
    char* ws = (char*)d_ws;
    size_t off = 0;
    auto walloc = [&](size_t bytes) { size_t r = off; off += (bytes + 255) & ~(size_t)255; return r; };
    size_t o_wts = walloc(4u  << 20);
    size_t o_zn  = walloc((size_t)NPAIR * CZD * 2);
    size_t o_f1  = walloc((size_t)NPAIR * CZD * 4);
    size_t o_f2  = walloc((size_t)NPAIR * CZD * 4);
    size_t o_f3  = walloc((size_t)NPAIR * CZD * 4);
    size_t o_ab  = walloc((size_t)NPAIR * CZD * 2);
    size_t o_bb  = walloc((size_t)NPAIR * CZD * 2);
    size_t o_cb  = walloc((size_t)NPAIR * CZD * 2);
    size_t o_tb  = walloc((size_t)NPAIR * HATT * 4);
    size_t o_t2  = walloc((size_t)NPAIR * TNCZ * 2);

    bf16*  zn   = (bf16*)(ws + o_zn);
    float* f1   = (float*)(ws + o_f1);
    float* f2   = (float*)(ws + o_f2);
    float* f3   = (float*)(ws + o_f3);
    bf16*  abuf = (bf16*)(ws + o_ab);
    bf16*  bbuf = (bf16*)(ws + o_bb);
    bf16*  cbuf = (bf16*)(ws + o_cb);
    float* tbb  = (float*)(ws + o_tb);
    bf16*  t1   = (bf16*)(ws + o_f1);     // transition t1 overlays f1+f2 (67MB)
    bf16*  t2   = (bf16*)(ws + o_t2);
    float* zcur = (float*)d_out;

    // ---------------- weight prep (f32 -> bf16 transposed) -----------------
    size_t woff = o_wts;
    auto prepW = [&](const float* w, int K, int Nout) -> bf16* {
        bf16* dst = (bf16*)(ws + woff);
        woff += (size_t)K * Nout * 2;
        int n = K * Nout;
        wprep_kernel<<<(n + 255) / 256, 256, 0, stream>>>(w, dst, K, Nout);
        return dst;
    };

    auto gemm = [&](const bf16* A, const bf16* Bt, const float* bias,
                    int M, int K, int Nout, int epi, float scale,
                    float* outF, bf16* outH, float* addOut,
                    long long ors, long long ocs, long long ooff, int tr,
                    int batch, long long aB, long long bB, long long oB) {
        dim3 g(Nout / 32, M / 64, batch);
        gemm_wmma_kernel<<<g, 256, 0, stream>>>(A, Bt, bias, M, K, Nout, epi, scale,
                                                outF, outH, addOut, ors, ocs, ooff, tr,
                                                aB, bB, oB);
    };
    auto lnk = [&](const float* src, const float* g, const float* b, bf16* dst, int tr) {
        ln_bf16_kernel<<<NPAIR / 8, 256, 0, stream>>>(src, g, b, dst, tr);
    };
    const int NE = NPAIR * CZD;

    // residual base: zcur = z  (device-to-device, graph-capture safe)
    hipMemcpyAsync(zcur, zin, (size_t)NE * 4, hipMemcpyDeviceToDevice, stream);

    // ---------------- TriangleMultiplicativeUpdate (x2) --------------------
    auto triMul = [&](const TriMulP& p, int outgoing) {
        bf16 *wag = prepW(p.ag_w, CZD, CZD), *wap = prepW(p.ap_w, CZD, CZD);
        bf16 *wbg = prepW(p.bg_w, CZD, CZD), *wbp = prepW(p.bp_w, CZD, CZD);
        bf16 *wo  = prepW(p.o_w,  CZD, CZD), *wg  = prepW(p.g_w,  CZD, CZD);
        lnk(zcur, p.ln_g, p.ln_b, zn, 0);
        int tr = outgoing ? 0 : 1;
        gemm(zn, wag, p.ag_b, NPAIR, CZD, CZD, 1, 1.f, f1, nullptr, nullptr, CZD, 1, 0, 0, 1, 0, 0, 0);
        gemm(zn, wap, p.ap_b, NPAIR, CZD, CZD, 0, 1.f, f2, nullptr, nullptr, CZD, 1, 0, 0, 1, 0, 0, 0);
        gatepack_kernel<<<(NE + 255) / 256, 256, 0, stream>>>(f1, f2, mask, abuf, tr);
        gemm(zn, wbg, p.bg_b, NPAIR, CZD, CZD, 1, 1.f, f1, nullptr, nullptr, CZD, 1, 0, 0, 1, 0, 0, 0);
        gemm(zn, wbp, p.bp_b, NPAIR, CZD, CZD, 0, 1.f, f2, nullptr, nullptr, CZD, 1, 0, 0, 1, 0, 0, 0);
        gatepack_kernel<<<(NE + 255) / 256, 256, 0, stream>>>(f1, f2, mask, bbuf, tr);
        // triangle einsum: 128 batched NT GEMMs, scatter to [i*256+j][c]
        gemm(abuf, bbuf, nullptr, NSEQ, NSEQ, NSEQ, 0, 1.f, f3, nullptr, nullptr,
             (long long)NSEQ * CZD, CZD, 0, 0, CZD, NPAIR, NPAIR, 1);
        lnk(f3, p.lno_g, p.lno_b, cbuf, 0);                       // out-LN -> bf16
        gemm(cbuf, wo, p.o_b, NPAIR, CZD, CZD, 0, 1.f, f2, nullptr, nullptr, CZD, 1, 0, 0, 1, 0, 0, 0);
        gemm(zn, wg, p.g_b, NPAIR, CZD, CZD, 1, 1.f, f1, nullptr, nullptr, CZD, 1, 0, 0, 1, 0, 0, 0);
        fma_add_kernel<<<(NE + 255) / 256, 256, 0, stream>>>(zcur, f2, f1, NE);
    };
    triMul(tmo, 1);
    triMul(tmi, 0);

    // ---------------- TriangleAttention (starting + ending) ----------------
    auto triAttn = [&](const TriAttnP& p, int transposed) {
        bf16 *wq = prepW(p.q_w, CZD, CZD), *wk = prepW(p.k_w, CZD, CZD);
        bf16 *wv = prepW(p.v_w, CZD, CZD), *wg = prepW(p.g_w, CZD, CZD);
        bf16 *wo = prepW(p.o_w, CZD, CZD);
        lnk(zcur, p.ln_g, p.ln_b, zn, transposed);
        const float qscale = 0.17677669529663688f;   // 1/sqrt(32)
        gemm(zn, wq, nullptr, NPAIR, CZD, CZD, 5, qscale, nullptr, abuf, nullptr, CZD, 1, 0, 0, 1, 0, 0, 0);
        gemm(zn, wk, nullptr, NPAIR, CZD, CZD, 2, 1.f, nullptr, bbuf, nullptr, CZD, 1, 0, 0, 1, 0, 0, 0);
        gemm(zn, wv, nullptr, NPAIR, CZD, CZD, 2, 1.f, nullptr, cbuf, nullptr, CZD, 1, 0, 0, 1, 0, 0, 0);
        tbproj_kernel<<<(NPAIR * HATT + 255) / 256, 256, 0, stream>>>(zn, p.tb_w, tbb);
        gemm(zn, wg, p.g_b, NPAIR, CZD, CZD, 1, 1.f, f1, nullptr, nullptr, CZD, 1, 0, 0, 1, 0, 0, 0);
        dim3 ag(NSEQ / 32, HATT, NSEQ);
        attn_kernel<<<ag, 128, 0, stream>>>(abuf, bbuf, cbuf, tbb, mask, transposed, f3);
        mul_f32_to_bf16_kernel<<<(NE + 255) / 256, 256, 0, stream>>>(f3, f1, abuf, NE);
        gemm(abuf, wo, p.o_b, NPAIR, CZD, CZD, 4, 1.f, nullptr, nullptr, zcur,
             CZD, 1, 0, transposed, 1, 0, 0, 0);
    };
    triAttn(tas, 0);
    triAttn(tae, 1);

    // ---------------- Transition (swish-gated MLP) -------------------------
    {
        bf16 *wa  = prepW(trs.a_w,   CZD,  TNCZ);
        bf16 *wb  = prepW(trs.b_w,   CZD,  TNCZ);
        bf16 *wout= prepW(trs.out_w, TNCZ, CZD);
        lnk(zcur, trs.ln_g, trs.ln_b, zn, 0);
        gemm(zn, wa, nullptr, NPAIR, CZD, TNCZ, 3, 1.f, nullptr, t1, nullptr, TNCZ, 1, 0, 0, 1, 0, 0, 0);
        gemm(zn, wb, nullptr, NPAIR, CZD, TNCZ, 2, 1.f, nullptr, t2, nullptr, TNCZ, 1, 0, 0, 1, 0, 0, 0);
        int n2 = NPAIR * TNCZ;
        mul_bf16_ip_kernel<<<(n2 + 255) / 256, 256, 0, stream>>>(t1, t2, n2);
        gemm(t1, wout, nullptr, NPAIR, TNCZ, CZD, 4, 1.f, nullptr, nullptr, zcur,
             CZD, 1, 0, 0, 1, 0, 0, 0);
    }
}
